// Matformer_18726057411347
// MI455X (gfx1250) — matrix operand order repaired and checked by reading, not compiled, not run on hardware
//
#include <hip/hip_runtime.h>
#include <stdint.h>

#define NN   10000
#define EE   100000
#define INF_ 128
#define CC   64
#define HH   4
#define HC   256     // H*C
#define C3   192     // 3*C
#define EPSF 1e-5f

typedef __bf16 bf16_t;
typedef __attribute__((ext_vector_type(16))) __bf16 v16bf;
typedef __attribute__((ext_vector_type(4)))  __bf16 v4bf;
typedef __attribute__((ext_vector_type(8)))  float  v8f;

__device__ __forceinline__ bf16_t f2bf(float f) { return (bf16_t)f; }
__device__ __forceinline__ float  bf2f(bf16_t b) { return (float)b; }

// A fragment: 16x32 bf16 (MxK). lane<16: row=lane, K 0..15 ; lane>=16: row=lane-16, K 16..31
__device__ __forceinline__ v16bf load_A_frag(const bf16_t* base, int lda) {
  int lane = threadIdx.x;
  int row  = lane & 15;
  int kb   = (lane < 16) ? 0 : 16;
  const bf16_t* p = base + (size_t)row * lda + kb;
  v16bf a;
#pragma unroll
  for (int i = 0; i < 8; i++) a[i] = p[i];
#pragma unroll
  for (int i = 0; i < 8; i++) a[8 + i] = p[8 + i];
  return a;
}

// Packed-B fragment: frag = 512 bf16 (1024B); lane's 16 bf16 contiguous at lane*16.
__device__ __forceinline__ v16bf load_B_packed(const bf16_t* frag) {
  return *(const v16bf*)(frag + threadIdx.x * 16);
}

#define WMMA_BF16(A, B, C) \
  __builtin_amdgcn_wmma_f32_16x16x32_bf16(false, (A), false, (B), (short)0, (C), false, false)

// ---------------------------------------------------------------- utility kernels
__global__ void zero_kernel(float* p, size_t n) {
  size_t i  = (size_t)blockIdx.x * blockDim.x + threadIdx.x;
  size_t st = (size_t)gridDim.x * blockDim.x;
  for (; i < n; i += st) p[i] = 0.f;
}

__global__ void cast_kernel(const float* __restrict__ in, bf16_t* __restrict__ o, size_t n) {
  size_t i  = (size_t)blockIdx.x * blockDim.x + threadIdx.x;
  size_t st = (size_t)gridDim.x * blockDim.x;
  for (; i < n; i += st) o[i] = f2bf(in[i]);
}

// Pack W[K,N] (f32, row-major) into WMMA B fragments (bf16):
// frag(kt,nt) at ((kt*(N/16))+nt)*512 elements; within frag, lane-major 16 elems each.
__global__ void pack_b_kernel(const float* __restrict__ W, bf16_t* __restrict__ out, int K, int N) {
  int t     = blockIdx.x * blockDim.x + threadIdx.x;
  int nt16  = N >> 4;
  int nfrag = (K >> 5) * nt16;
  if (t >= nfrag * 32) return;
  int lane = t & 31;
  int f    = t >> 5;
  int kt   = f / nt16, nt = f - kt * nt16;
  int col  = nt * 16 + (lane & 15);
  int kb   = kt * 32 + ((lane < 16) ? 0 : 16);
  bf16_t* o = out + (size_t)f * 512 + lane * 16;
#pragma unroll
  for (int i = 0; i < 16; i++) o[i] = f2bf(W[(size_t)(kb + i) * N + col]);
}

// ---------------------------------------------------------------- generic WMMA GEMM
// C[M,N] = A[M,K](bf16) @ Bp(packed bf16) + bias.
// block=(32,4): 4 waves = 4 row tiles; each wave computes 4 col tiles (A frag reused 4x).
// Requires N multiple of 64, K multiple of 32.
template <bool BF16_OUT>
__global__ void gemm_bias_kernel(const bf16_t* __restrict__ A, const bf16_t* __restrict__ Bp,
                                 const float* __restrict__ bias, void* __restrict__ Cout,
                                 int M, int N, int K) {
  int tm = blockIdx.y * blockDim.y + threadIdx.y;  // row tile
  if (tm * 16 >= M) return;                        // wave-uniform
  int nt16 = N >> 4;
  int nt0  = blockIdx.x * 4;                       // first of 4 col tiles
  const bf16_t* Arow = A + (size_t)tm * 16 * K;
  v8f acc[4] = {};
  for (int k0 = 0; k0 < K; k0 += 32) {
    const bf16_t* fb = Bp + ((size_t)(k0 >> 5) * nt16 + nt0) * 512;
    if (k0 + 32 < K) __builtin_prefetch(fb + (size_t)nt16 * 512, 0, 1);
    v16bf af = load_A_frag(Arow + k0, K);
#pragma unroll
    for (int u = 0; u < 4; u++) {
      v16bf bf = load_B_packed(fb + u * 512);
      acc[u] = WMMA_BF16(af, bf, acc[u]);
    }
  }
  int lane = threadIdx.x;
  int col  = lane & 15;
  int rb   = (lane < 16) ? 0 : 8;
#pragma unroll
  for (int u = 0; u < 4; u++) {
    int   gcol = (nt0 + u) * 16 + col;
    float bvv  = bias ? bias[gcol] : 0.f;
#pragma unroll
    for (int i = 0; i < 8; i++) {
      int   grow = tm * 16 + rb + i;
      float vv   = acc[u][i] + bvv;
      if (BF16_OUT)
        ((bf16_t*)Cout)[(size_t)grow * N + gcol] = f2bf(vv);
      else
        ((float*)Cout)[(size_t)grow * N + gcol] = vv;
    }
  }
}

// ---------------------------------------------------------------- fused edge kernel
// Block = 16 edges x 4 heads = 64 rows of 192. 256 threads (8 waves).
__global__ void __launch_bounds__(256) edge_kernel(
    const long long* __restrict__ ei,                                     // [2,E]
    const float* __restrict__ q, const float* __restrict__ k, const float* __restrict__ v,  // [N,256]
    const bf16_t* __restrict__ e_bf,                                      // [E,256]
    const bf16_t* __restrict__ Wmsg, const float* __restrict__ bmsg,      // packed [192,192],[192]
    const bf16_t* __restrict__ Wml, const float* __restrict__ bml,        // packed [192,64],[64]
    const float* __restrict__ g_ln3, const float* __restrict__ b_ln3,     // [192]
    const float* __restrict__ g_ml, const float* __restrict__ b_ml,       // [64]
    float* __restrict__ agg)                                              // [N,256]
{
  __shared__ int    s_src[16], s_dst[16];
  __shared__ bf16_t s_m[64 * C3];     // m_in, then gated m1
  __shared__ bf16_t s_gate[64 * C3];  // sigmoid gate
  __shared__ float  s_rsum[64 * 2];
  __shared__ float  s_rsq[64 * 2];

  const int lane = threadIdx.x;
  const int wv   = threadIdx.y;
  const int tid  = wv * 32 + lane;

  if (tid < 16) {
    s_src[tid] = (int)ei[(size_t)blockIdx.x * 16 + tid];
    s_dst[tid] = (int)ei[(size_t)EE + (size_t)blockIdx.x * 16 + tid];
  }
  __syncthreads();

  // ---- Phase 1: alpha -> LN(192) -> sigmoid gate ; build m_in (bf16) in LDS
  {
    int r   = tid >> 2;   // edge-head row 0..63
    int sub = tid & 3;    // 4 threads per row, 48 cols each
    int c0  = sub * 48;
    int el  = r >> 2;
    int h   = r & 3;
    int src = s_src[el], dst = s_dst[el];
    size_t qb  = (size_t)dst * HC + h * CC;
    size_t kbj = (size_t)src * HC + h * CC;
    size_t ebE = ((size_t)blockIdx.x * 16 + el) * HC + h * CC;
    const float rs3c = 0.07216878364870322f;  // 1/sqrt(192)
    float av[48];
    float sum = 0.f, sq = 0.f;
#pragma unroll
    for (int j4 = 0; j4 < 48; j4 += 4) {   // aligned groups of 4 never straddle segments
      int    c  = c0 + j4;
      int    cc = c & 63;
      float4 qv = *(const float4*)(q + qb + cc);
      float4 kv, mv;
      if (c < 64) {
        kv = *(const float4*)(k + qb + cc);
        mv = *(const float4*)(v + qb + cc);
      } else if (c < 128) {
        kv = *(const float4*)(k + kbj + cc);
        mv = *(const float4*)(v + kbj + cc);
      } else {
        v4bf ev = *(const v4bf*)(e_bf + ebE + cc);
        kv = make_float4(bf2f(ev[0]), bf2f(ev[1]), bf2f(ev[2]), bf2f(ev[3]));
        mv = kv;
      }
      float a0 = qv.x * kv.x * rs3c, a1 = qv.y * kv.y * rs3c;
      float a2 = qv.z * kv.z * rs3c, a3 = qv.w * kv.w * rs3c;
      av[j4] = a0; av[j4 + 1] = a1; av[j4 + 2] = a2; av[j4 + 3] = a3;
      sum += a0 + a1 + a2 + a3;
      sq  += a0 * a0 + a1 * a1 + a2 * a2 + a3 * a3;
      v4bf ms;
      ms[0] = f2bf(mv.x); ms[1] = f2bf(mv.y); ms[2] = f2bf(mv.z); ms[3] = f2bf(mv.w);
      *(v4bf*)&s_m[r * C3 + c] = ms;
    }
    // reduce across the aligned quad of lanes owning this row
    sum += __shfl_xor(sum, 1); sum += __shfl_xor(sum, 2);
    sq  += __shfl_xor(sq, 1);  sq  += __shfl_xor(sq, 2);
    float mean = sum * (1.f / C3);
    float var  = sq * (1.f / C3) - mean * mean;
    float rstd = rsqrtf(var + EPSF);
#pragma unroll
    for (int j4 = 0; j4 < 48; j4 += 4) {
      int    c  = c0 + j4;
      float4 gl = *(const float4*)(g_ln3 + c);
      float4 bl = *(const float4*)(b_ln3 + c);
      v4bf gs;
      gs[0] = f2bf(1.f / (1.f + __expf(-(gl.x * (av[j4]     - mean) * rstd + bl.x))));
      gs[1] = f2bf(1.f / (1.f + __expf(-(gl.y * (av[j4 + 1] - mean) * rstd + bl.y))));
      gs[2] = f2bf(1.f / (1.f + __expf(-(gl.z * (av[j4 + 2] - mean) * rstd + bl.z))));
      gs[3] = f2bf(1.f / (1.f + __expf(-(gl.w * (av[j4 + 3] - mean) * rstd + bl.w))));
      *(v4bf*)&s_gate[r * C3 + c] = gs;
    }
  }
  __syncthreads();

  // ---- Phase 2: m1[64,192] = m_in @ Wmsg + bmsg, gated (WMMA, LDS A, packed-B)
  {
    int rt   = wv >> 1;  // row-tile 0..3
    int half = wv & 1;   // 96-col half
    v8f acc1[6] = {};
    for (int k0 = 0; k0 < C3; k0 += 32) {
      v16bf af = load_A_frag(&s_m[(rt * 16) * C3 + k0], C3);
      const bf16_t* fb = Wmsg + ((size_t)(k0 >> 5) * 12 + 6 * half) * 512;
#pragma unroll
      for (int u = 0; u < 6; u++) {
        v16bf bf = load_B_packed(fb + u * 512);
        acc1[u] = WMMA_BF16(af, bf, acc1[u]);
      }
    }
    __syncthreads();  // everyone done reading m_in; safe to overwrite
    int rb = (lane < 16) ? 0 : 8;
#pragma unroll
    for (int u = 0; u < 6; u++) {
      int   colb = 96 * half + 16 * u + (lane & 15);
      float bb   = bmsg[colb];
#pragma unroll
      for (int i = 0; i < 8; i++) {
        int   row = rt * 16 + rb + i;
        float g   = bf2f(s_gate[row * C3 + colb]);
        s_m[row * C3 + colb] = f2bf((acc1[u][i] + bb) * g);
      }
    }
  }
  __syncthreads();

  // ---- Phase 3: m2[64,64] = m1g @ Wml + bml ; LN(64) ; scatter-add to agg[dst]
  {
    int rt   = wv >> 1;
    int half = wv & 1;   // 32-col half
    v8f acc2[2] = {};
    for (int k0 = 0; k0 < C3; k0 += 32) {
      v16bf af = load_A_frag(&s_m[(rt * 16) * C3 + k0], C3);
      const bf16_t* fb = Wml + ((size_t)(k0 >> 5) * 4 + 2 * half) * 512;
#pragma unroll
      for (int t = 0; t < 2; t++) {
        v16bf bf = load_B_packed(fb + t * 512);
        acc2[t] = WMMA_BF16(af, bf, acc2[t]);
      }
    }
    int rb = (lane < 16) ? 0 : 8;
    float psum[8], psq[8];
#pragma unroll
    for (int i = 0; i < 8; i++) { psum[i] = 0.f; psq[i] = 0.f; }
#pragma unroll
    for (int t = 0; t < 2; t++) {
      int   chan = 32 * half + 16 * t + (lane & 15);
      float bb   = bml[chan];
#pragma unroll
      for (int i = 0; i < 8; i++) {
        acc2[t][i] += bb;
        psum[i] += acc2[t][i];
        psq[i]  += acc2[t][i] * acc2[t][i];
      }
    }
#pragma unroll
    for (int i = 0; i < 8; i++) {
#pragma unroll
      for (int o = 1; o < 16; o <<= 1) {   // independent half-wave reductions
        psum[i] += __shfl_xor(psum[i], o);
        psq[i]  += __shfl_xor(psq[i], o);
      }
      if ((lane & 15) == 0) {
        int row = rt * 16 + rb + i;
        s_rsum[row * 2 + half] = psum[i];
        s_rsq[row * 2 + half]  = psq[i];
      }
    }
    __syncthreads();
#pragma unroll
    for (int t = 0; t < 2; t++) {
      int   chan = 32 * half + 16 * t + (lane & 15);
      float gch = g_ml[chan], bch = b_ml[chan];
#pragma unroll
      for (int i = 0; i < 8; i++) {
        int   row  = rt * 16 + rb + i;
        float mean = (s_rsum[row * 2] + s_rsum[row * 2 + 1]) * (1.f / CC);
        float var  = (s_rsq[row * 2] + s_rsq[row * 2 + 1]) * (1.f / CC) - mean * mean;
        float rstd = rsqrtf(var + EPSF);
        float nv   = gch * (acc2[t][i] - mean) * rstd + bch;
        int   el = row >> 2, h = row & 3;
        atomicAdd(&agg[(size_t)s_dst[el] * HC + h * CC + chan], nv);
      }
    }
  }
}

// ---------------------------------------------------------------- batchnorm epilogue
__global__ void bn_stats_kernel(const float* __restrict__ out0, float* __restrict__ bns, int Nr) {
  int   c = threadIdx.x;  // 64 threads, one per channel
  float s = 0.f, q2 = 0.f;
  int r0 = blockIdx.x * 100;
  int r1 = r0 + 100; if (r1 > Nr) r1 = Nr;
  for (int r = r0; r < r1; ++r) {
    float xv = out0[(size_t)r * CC + c];
    s += xv; q2 += xv * xv;
  }
  atomicAdd(&bns[c], s);
  atomicAdd(&bns[64 + c], q2);
}

__global__ void final_kernel(const float* __restrict__ out0, const float* __restrict__ skip,
                             const float* __restrict__ bns, const float* __restrict__ g_bn,
                             const float* __restrict__ b_bn, float* __restrict__ out, int Nr) {
  int idx = blockIdx.x * blockDim.x + threadIdx.x;
  if (idx >= Nr * CC) return;
  int   c    = idx & 63;
  float invn = 1.f / (float)Nr;
  float mean = bns[c] * invn;
  float var  = bns[64 + c] * invn - mean * mean;
  float xv   = out0[idx];
  float bn   = g_bn[c] * (xv - mean) * rsqrtf(var + EPSF) + b_bn[c];
  float si   = bn / (1.f + __expf(-bn));
  out[idx] = si + skip[idx];
}

// ---------------------------------------------------------------- launcher
extern "C" void kernel_launch(void* const* d_in, const int* in_sizes, int n_in,
                              void* d_out, int out_size, void* d_ws, size_t ws_size,
                              hipStream_t stream) {
  (void)in_sizes; (void)n_in; (void)out_size; (void)ws_size;
  const float*     x  = (const float*)d_in[0];
  const long long* ei = (const long long*)d_in[1];
  const float*     ea = (const float*)d_in[2];
  const float* Wq = (const float*)d_in[3];   const float* bq = (const float*)d_in[4];
  const float* Wk = (const float*)d_in[5];   const float* bk = (const float*)d_in[6];
  const float* Wv = (const float*)d_in[7];   const float* bv = (const float*)d_in[8];
  const float* We = (const float*)d_in[9];
  const float* Wskip = (const float*)d_in[10]; const float* bskip = (const float*)d_in[11];
  const float* Wcat  = (const float*)d_in[12]; const float* bcat  = (const float*)d_in[13];
  const float* Wmsg  = (const float*)d_in[14]; const float* bmsg  = (const float*)d_in[15];
  const float* Wml   = (const float*)d_in[16]; const float* bml   = (const float*)d_in[17];
  const float* g_ml  = (const float*)d_in[18]; const float* b_ml  = (const float*)d_in[19];
  const float* g_ln3 = (const float*)d_in[20]; const float* b_ln3 = (const float*)d_in[21];
  const float* g_bn  = (const float*)d_in[22]; const float* b_bn  = (const float*)d_in[23];
  float* out = (float*)d_out;

  char*  ws  = (char*)d_ws;
  size_t off = 0;
  auto alloc = [&](size_t bytes) -> char* {
    char* p = ws + off;
    off += (bytes + 255) & ~(size_t)255;
    return p;
  };

  bf16_t* x_bf     = (bf16_t*)alloc((size_t)NN * INF_ * 2);
  bf16_t* ea_bf    = (bf16_t*)alloc((size_t)EE * INF_ * 2);
  bf16_t* wq_p     = (bf16_t*)alloc((size_t)INF_ * HC * 2);
  bf16_t* wk_p     = (bf16_t*)alloc((size_t)INF_ * HC * 2);
  bf16_t* wv_p     = (bf16_t*)alloc((size_t)INF_ * HC * 2);
  bf16_t* we_p     = (bf16_t*)alloc((size_t)INF_ * HC * 2);
  bf16_t* wskip_p  = (bf16_t*)alloc((size_t)INF_ * CC * 2);
  bf16_t* wcat_p   = (bf16_t*)alloc((size_t)HC * CC * 2);
  bf16_t* wmsg_p   = (bf16_t*)alloc((size_t)C3 * C3 * 2);
  bf16_t* wml_p    = (bf16_t*)alloc((size_t)C3 * CC * 2);
  float*  qf       = (float*)alloc((size_t)NN * HC * 4);
  float*  kf       = (float*)alloc((size_t)NN * HC * 4);
  float*  vf       = (float*)alloc((size_t)NN * HC * 4);
  bf16_t* e_bf     = (bf16_t*)alloc((size_t)EE * HC * 2);
  float*  skip     = (float*)alloc((size_t)NN * CC * 4);
  float*  agg      = (float*)alloc((size_t)NN * HC * 4);
  bf16_t* agg_bf   = (bf16_t*)alloc((size_t)NN * HC * 2);
  float*  out0     = (float*)alloc((size_t)NN * CC * 4);
  float*  bns      = (float*)alloc(128 * 4);

  auto cast = [&](const float* in, bf16_t* o, size_t n) {
    unsigned g = (unsigned)((n + 255) / 256);
    if (g > 65535u) g = 65535u;
    cast_kernel<<<g, 256, 0, stream>>>(in, o, n);
  };
  auto packb = [&](const float* W, bf16_t* o, int K, int N) {
    int threads = (K >> 5) * (N >> 4) * 32;
    pack_b_kernel<<<(threads + 255) / 256, 256, 0, stream>>>(W, o, K, N);
  };

  cast(x, x_bf, (size_t)NN * INF_);
  cast(ea, ea_bf, (size_t)EE * INF_);
  packb(Wq, wq_p, INF_, HC);
  packb(Wk, wk_p, INF_, HC);
  packb(Wv, wv_p, INF_, HC);
  packb(We, we_p, INF_, HC);
  packb(Wskip, wskip_p, INF_, CC);
  packb(Wcat, wcat_p, HC, CC);
  packb(Wmsg, wmsg_p, C3, C3);
  packb(Wml, wml_p, C3, CC);

  zero_kernel<<<512, 256, 0, stream>>>(agg, (size_t)NN * HC);
  zero_kernel<<<1, 128, 0, stream>>>(bns, 128);

  dim3 gb(32, 4);
  int  mty_n = (NN / 16 + 3) / 4;  // 157 row-tile groups for M=10000
  int  mty_e = (EE / 16 + 3) / 4;  // 1563 for M=100000
  // q/k/v projections: [N,128] @ [128,256]
  gemm_bias_kernel<false><<<dim3(HC / 64, mty_n), gb, 0, stream>>>(x_bf, wq_p, bq, qf, NN, HC, INF_);
  gemm_bias_kernel<false><<<dim3(HC / 64, mty_n), gb, 0, stream>>>(x_bf, wk_p, bk, kf, NN, HC, INF_);
  gemm_bias_kernel<false><<<dim3(HC / 64, mty_n), gb, 0, stream>>>(x_bf, wv_p, bv, vf, NN, HC, INF_);
  // skip: [N,128] @ [128,64]
  gemm_bias_kernel<false><<<dim3(1, mty_n), gb, 0, stream>>>(x_bf, wskip_p, bskip, skip, NN, CC, INF_);
  // e: [E,128] @ [128,256] -> bf16
  gemm_bias_kernel<true><<<dim3(HC / 64, mty_e), gb, 0, stream>>>(ea_bf, we_p, nullptr, e_bf, EE, HC, INF_);

  // fused per-edge attention / message / LN / scatter
  edge_kernel<<<EE / 16, dim3(32, 8), 0, stream>>>(ei, qf, kf, vf, e_bf,
                                                   wmsg_p, bmsg, wml_p, bml,
                                                   g_ln3, b_ln3, g_ml, b_ml, agg);

  // output projection: agg[N,256] @ Wcat[256,64]
  cast(agg, agg_bf, (size_t)NN * HC);
  gemm_bias_kernel<false><<<dim3(1, mty_n), gb, 0, stream>>>(agg_bf, wcat_p, bcat, out0, NN, CC, HC);

  // batchnorm stats + final elementwise
  bn_stats_kernel<<<(NN + 99) / 100, 64, 0, stream>>>(out0, bns, NN);
  final_kernel<<<(NN * CC + 255) / 256, 256, 0, stream>>>(out0, skip, bns, g_bn, b_bn, out, NN);
}
